// ScaledDotProductAttention_88046829568797
// MI455X (gfx1250) — matrix operand order repaired and checked by reading, not compiled, not run on hardware
//
#include <hip/hip_runtime.h>

typedef __attribute__((ext_vector_type(16))) _Float16 v16h;
typedef __attribute__((ext_vector_type(8)))  _Float16 v8h;
typedef __attribute__((ext_vector_type(4)))  _Float16 v4h;
typedef __attribute__((ext_vector_type(2)))  __fp16   fp16x2;
typedef __attribute__((ext_vector_type(8)))  float    v8f;

#define B_    2
#define H_    16
#define S_    2048
#define D_    64

#define QTILE 64            // queries per block (4 waves x 16)
#define KTILE 32            // keys per main-loop iteration (= WMMA K dim for PV)
#define NTILES (S_ / KTILE)
#define NTHREADS 128
#define KPAD  (KTILE + 8)
#define DPAD  (D_ + 8)

// Transposed formulation: S^T = K.Q^T (keys=M, queries=N), O^T = V^T.P^T.
// Each lane owns ONE query (n = lane&15) and 16 keys in two runs of 8, so the
// softmax state is a single (m,l) pair per lane and mask bytes form two contiguous 8-byte runs.

// byte -> float; ISel folds (uitofp (and (srl w, 8r), 0xff)) to v_cvt_f32_ubyteN
__device__ __forceinline__ float ubyte_to_f32(unsigned int w, int bsel) {
    return (float)((w >> (8 * (bsel & 3))) & 0xffu);
}

__global__ __launch_bounds__(NTHREADS)
void fa_fwd_kernel(const float* __restrict__ Q, const float* __restrict__ K,
                   const float* __restrict__ V, const unsigned char* __restrict__ Mask,
                   float* __restrict__ O)
{
    __shared__ _Float16 sQ [QTILE][DPAD];       // queries x d (row-major)
    __shared__ _Float16 sK [2][KTILE][DPAD];    // keys x d (row-major), double buffered
    __shared__ _Float16 sVt[2][D_][KPAD];       // d x keys (transposed), double buffered
    __shared__ _Float16 sPt[4][16][KPAD];       // per-wave P (query x key) scratch

    const int tid  = threadIdx.x;
    const int wave = tid >> 5;
    const int lane = tid & 31;
    const int bh   = blockIdx.y;                // b*H + h
    const int b    = bh / H_;
    const int qb   = blockIdx.x * QTILE;
    const int qw   = qb + wave * 16;            // this wave's query base row

    const int n    = lane & 15;                 // query index (N dim) owned by this lane
    const int kb16 = (lane >> 4) << 4;          // K offset 0 or 16, contiguous order (unused)
    const int rb8  = (lane >> 4) << 3;          // C/D M offset 0/8 (= A-layout K half offset)

    // softmax in exp2 domain: fold 1/sqrt(64) * log2(e) into the score scale
    const float SCALE2 = 0.125f * 1.4426950408889634f;

    const float* Qb = Q + (size_t)bh * S_ * D_;
    const float* Kb = K + (size_t)bh * S_ * D_;
    const float* Vb = V + (size_t)bh * S_ * D_;
    const unsigned char* Mb = Mask + (size_t)b * S_ * S_;

    // ---- stage Q tile once: 64x64 fp32 -> f16 in LDS ----
    {
        const float4* src = (const float4*)(Qb + (size_t)qb * D_);   // contiguous tile
        for (int i = tid; i < QTILE * D_ / 4; i += NTHREADS) {
            int row = i >> 4;                   // D_/4 == 16 float4 per row
            int col = (i & 15) << 2;
            float4 f = src[i];
            fp16x2 a = __builtin_amdgcn_cvt_pkrtz(f.x, f.y);
            fp16x2 c = __builtin_amdgcn_cvt_pkrtz(f.z, f.w);
            v4h q4;
            q4[0] = (_Float16)a.x; q4[1] = (_Float16)a.y;
            q4[2] = (_Float16)c.x; q4[3] = (_Float16)c.y;
            *(v4h*)&sQ[row][col] = q4;
        }
    }

    // ---- cooperative K/V tile loader: K row-major f16 (packed), V transposed ----
    auto load_tile = [&](int buf, int kt) {
        const float4* ks = (const float4*)(Kb + (size_t)kt * D_);
        const float4* vs = (const float4*)(Vb + (size_t)kt * D_);
#pragma unroll
        for (int ii = 0; ii < (KTILE * D_ / 4) / NTHREADS; ++ii) {
            int i   = tid + ii * NTHREADS;
            int row = i >> 4;                   // key index 0..31
            int col = (i & 15) << 2;            // d index
            float4 f = ks[i];
            fp16x2 a = __builtin_amdgcn_cvt_pkrtz(f.x, f.y);
            fp16x2 c = __builtin_amdgcn_cvt_pkrtz(f.z, f.w);
            v4h k4;
            k4[0] = (_Float16)a.x; k4[1] = (_Float16)a.y;
            k4[2] = (_Float16)c.x; k4[3] = (_Float16)c.y;
            *(v4h*)&sK[buf][row][col] = k4;
            float4 g = vs[i];                   // transposed scatter: elementwise
            sVt[buf][col + 0][row] = (_Float16)g.x;
            sVt[buf][col + 1][row] = (_Float16)g.y;
            sVt[buf][col + 2][row] = (_Float16)g.z;
            sVt[buf][col + 3][row] = (_Float16)g.w;
        }
    };

    load_tile(0, 0);
    __syncthreads();

    // ---- loop-invariant Q^T B-fragments: B[e] = Q[qw+n][32c + rb8 + e + 8*(e/8)] ----
    v16h bQ0, bQ1;
#pragma unroll
    for (int e = 0; e < 16; ++e) {
        bQ0[e] = sQ[(wave << 4) + n][rb8 + e + ((e >> 3) << 3)];
        bQ1[e] = sQ[(wave << 4) + n][32 + rb8 + e + ((e >> 3) << 3)];
    }

    // ---- persistent state: O^T accumulators + per-lane (m,l) ----
    v8f acc0 = {}, acc1 = {}, acc2 = {}, acc3 = {};
    float mrun = -3.0e38f, lrun = 0.0f;

    const unsigned char* mrow = Mb + (size_t)(qw + n) * S_;   // this lane's mask row

    for (int t = 0; t < NTILES; ++t) {
        const int kt  = t * KTILE;
        const int cur = t & 1;

        // issue next tile's global loads first so they overlap this tile's compute
        if (t + 1 < NTILES) {
            load_tile(cur ^ 1, kt + KTILE);
            if (t + 2 < NTILES) {
                __builtin_prefetch(Kb + (size_t)(kt + 2 * KTILE) * D_ + tid * 16, 0, 1);
                __builtin_prefetch(Vb + (size_t)(kt + 2 * KTILE) * D_ + tid * 16, 0, 1);
            }
        }

        // ---- S^T = K Q^T : two 16x16 tiles (key groups as M) ----
        v16h aK00, aK01, aK10, aK11;
#pragma unroll
        for (int e = 0; e < 16; ++e) {
            int v  = e >> 1;
            int kk = ((v >> 2) << 4) + rb8 + ((v & 3) << 1) + (e & 1);   // d within 32-chunk
            int m  = lane & 15;
            aK00[e] = sK[cur][m][kk];              // keys 0-15, d 0-31
            aK01[e] = sK[cur][m][32 + kk];         // keys 0-15, d 32-63
            aK10[e] = sK[cur][16 + m][kk];         // keys 16-31, d 0-31
            aK11[e] = sK[cur][16 + m][32 + kk];    // keys 16-31, d 32-63
        }
        v8f s0 = {}, s1 = {};
        s0 = __builtin_amdgcn_wmma_f32_16x16x32_f16(false, aK00, false, bQ0, (short)0, s0, false, false);
        s0 = __builtin_amdgcn_wmma_f32_16x16x32_f16(false, aK01, false, bQ1, (short)0, s0, false, false);
        s1 = __builtin_amdgcn_wmma_f32_16x16x32_f16(false, aK10, false, bQ0, (short)0, s1, false, false);
        s1 = __builtin_amdgcn_wmma_f32_16x16x32_f16(false, aK11, false, bQ1, (short)0, s1, false, false);
        // lane holds, for query qw+n: keys {rb8+0..7} (s0) and {16+rb8+0..7} (s1)

        // ---- mask: two contiguous 8-byte runs of this lane's mask row ----
        const uint2 mw0 = *(const uint2*)(mrow + kt + rb8);
        const uint2 mw1 = *(const uint2*)(mrow + kt + 16 + rb8);

        // x = s*scale - 1e9 * mask_byte  (v_cvt_f32_ubyteN + v_fma per element)
        float x[16];
#pragma unroll
        for (int r = 0; r < 8; ++r) {
            unsigned int w0 = (r < 4) ? mw0.x : mw0.y;
            unsigned int w1 = (r < 4) ? mw1.x : mw1.y;
            x[r]     = fmaf(ubyte_to_f32(w0, r), -1.0e9f, s0[r] * SCALE2);
            x[8 + r] = fmaf(ubyte_to_f32(w1, r), -1.0e9f, s1[r] * SCALE2);
        }

        // ---- online softmax: in-lane tree + one partner-lane exchange ----
        float mx01 = fmaxf(x[0], x[1]),   mx23 = fmaxf(x[2], x[3]);
        float mx45 = fmaxf(x[4], x[5]),   mx67 = fmaxf(x[6], x[7]);
        float mx89 = fmaxf(x[8], x[9]),   mxab = fmaxf(x[10], x[11]);
        float mxcd = fmaxf(x[12], x[13]), mxef = fmaxf(x[14], x[15]);
        float mx = fmaxf(fmaxf(fmaxf(mx01, mx23), fmaxf(mx45, mx67)),
                         fmaxf(fmaxf(mx89, mxab), fmaxf(mxcd, mxef)));
        mx = fmaxf(mx, __shfl_xor(mx, 16, 32));      // partner lane owns other 16 keys

        float mnew  = fmaxf(mrun, mx);
        float alpha = exp2f(mrun - mnew);

        float p[16], rs = 0.0f;
#pragma unroll
        for (int j = 0; j < 16; ++j) { p[j] = exp2f(x[j] - mnew); rs += p[j]; }
        rs += __shfl_xor(rs, 16, 32);

        mrun = mnew;
        lrun = lrun * alpha + rs;
#pragma unroll
        for (int r = 0; r < 8; ++r) {
            acc0[r] *= alpha; acc1[r] *= alpha; acc2[r] *= alpha; acc3[r] *= alpha;
        }

        // ---- pack P to f16 pairs, 2x b128 stores into per-wave scratch ----
        {
            v8h plo, phi;
#pragma unroll
            for (int i = 0; i < 4; ++i) {
                fp16x2 a = __builtin_amdgcn_cvt_pkrtz(p[2 * i],     p[2 * i + 1]);
                fp16x2 c = __builtin_amdgcn_cvt_pkrtz(p[8 + 2 * i], p[8 + 2 * i + 1]);
                plo[2 * i] = (_Float16)a.x; plo[2 * i + 1] = (_Float16)a.y;
                phi[2 * i] = (_Float16)c.x; phi[2 * i + 1] = (_Float16)c.y;
            }
            *(v8h*)&sPt[wave][n][rb8]      = plo;   // keys rb8..rb8+7
            *(v8h*)&sPt[wave][n][16 + rb8] = phi;   // keys 16+rb8..+7
        }

        // ---- O^T += V^T P^T : A = V^T d-groups, B = P^T (contiguous reads) ----
        v16h bP;
#pragma unroll
        for (int e = 0; e < 16; ++e) bP[e] = sPt[wave][n][rb8 + e + ((e >> 3) << 3)];  // intra-wave RAW, in-order

        v16h aV0, aV1, aV2, aV3;
#pragma unroll
        for (int e = 0; e < 16; ++e) {
            int v  = e >> 1;
            int kk = ((v >> 2) << 4) + rb8 + ((v & 3) << 1) + (e & 1);   // key within 32
            int m  = lane & 15;
            aV0[e] = sVt[cur][m][kk];
            aV1[e] = sVt[cur][16 + m][kk];
            aV2[e] = sVt[cur][32 + m][kk];
            aV3[e] = sVt[cur][48 + m][kk];
        }
        acc0 = __builtin_amdgcn_wmma_f32_16x16x32_f16(false, aV0, false, bP, (short)0, acc0, false, false);
        acc1 = __builtin_amdgcn_wmma_f32_16x16x32_f16(false, aV1, false, bP, (short)0, acc1, false, false);
        acc2 = __builtin_amdgcn_wmma_f32_16x16x32_f16(false, aV2, false, bP, (short)0, acc2, false, false);
        acc3 = __builtin_amdgcn_wmma_f32_16x16x32_f16(false, aV3, false, bP, (short)0, acc3, false, false);

        // one barrier per iteration: next buffer visible, current buffer reusable
        __syncthreads();
    }

    // ---- epilogue: O[q][d] = acc^T / l ; 8 consecutive d per lane -> b128 stores ----
    {
        float inv = 1.0f / lrun;
        float* orow = O + ((size_t)bh * S_ + (qw + n)) * D_;
        float4 o;
        o.x = acc0[0] * inv; o.y = acc0[1] * inv; o.z = acc0[2] * inv; o.w = acc0[3] * inv;
        *(float4*)(orow + 0  + rb8)     = o;
        o.x = acc0[4] * inv; o.y = acc0[5] * inv; o.z = acc0[6] * inv; o.w = acc0[7] * inv;
        *(float4*)(orow + 0  + rb8 + 4) = o;
        o.x = acc1[0] * inv; o.y = acc1[1] * inv; o.z = acc1[2] * inv; o.w = acc1[3] * inv;
        *(float4*)(orow + 16 + rb8)     = o;
        o.x = acc1[4] * inv; o.y = acc1[5] * inv; o.z = acc1[6] * inv; o.w = acc1[7] * inv;
        *(float4*)(orow + 16 + rb8 + 4) = o;
        o.x = acc2[0] * inv; o.y = acc2[1] * inv; o.z = acc2[2] * inv; o.w = acc2[3] * inv;
        *(float4*)(orow + 32 + rb8)     = o;
        o.x = acc2[4] * inv; o.y = acc2[5] * inv; o.z = acc2[6] * inv; o.w = acc2[7] * inv;
        *(float4*)(orow + 32 + rb8 + 4) = o;
        o.x = acc3[0] * inv; o.y = acc3[1] * inv; o.z = acc3[2] * inv; o.w = acc3[3] * inv;
        *(float4*)(orow + 48 + rb8)     = o;
        o.x = acc3[4] * inv; o.y = acc3[5] * inv; o.z = acc3[6] * inv; o.w = acc3[7] * inv;
        *(float4*)(orow + 48 + rb8 + 4) = o;
    }
}

extern "C" void kernel_launch(void* const* d_in, const int* in_sizes, int n_in,
                              void* d_out, int out_size, void* d_ws, size_t ws_size,
                              hipStream_t stream)
{
    (void)in_sizes; (void)n_in; (void)out_size; (void)d_ws; (void)ws_size;
    const float* Q = (const float*)d_in[0];
    const float* K = (const float*)d_in[1];
    const float* V = (const float*)d_in[2];
    const unsigned char* M = (const unsigned char*)d_in[3];   // jnp.bool_ -> 1 byte/elt
    float* O = (float*)d_out;

    dim3 grid(S_ / QTILE, B_ * H_);   // 32 x 32 = 1024 workgroups
    dim3 block(NTHREADS);             // 4 waves (wave32)
    fa_fwd_kernel<<<grid, block, 0, stream>>>(Q, K, V, M, O);
}
